// Context2Query_77283641524595
// MI455X (gfx1250) — matrix operand order repaired and checked by reading, not compiled, not run on hardware
//
#include <hip/hip_runtime.h>
#include <hip/hip_bf16.h>

// Problem constants (fixed by the reference).
#define T_DIM 8192
#define J_DIM 2048
#define D_DIM 2048

// GEMM tiling: 128x128 workgroup tile, 8 waves, 32x64 per wave (2x4 WMMA accums).
#define BM 128
#define BN 128
#define BK 32
#define LDA 40   // padded K-stride (bf16 units) for A tile rows
#define LDB 40   // padded K-stride (bf16 units) for B tile columns (K-major)

typedef __attribute__((ext_vector_type(16))) __bf16 v16bf;
typedef __attribute__((ext_vector_type(8)))  float  v8f;

__device__ __forceinline__ unsigned int f32_to_bf16_bits(float f) {
    union { float f; unsigned int u; } c; c.f = f;
    unsigned int u = c.u;
    return (u + 0x7FFFu + ((u >> 16) & 1u)) >> 16;   // round-to-nearest-even
}

// Async DMA: global memory -> LDS, 16 bytes per lane, tracked by ASYNCcnt.
__device__ __forceinline__ void async_copy_b128(unsigned lds_byte_addr,
                                                const void* gaddr) {
    asm volatile("global_load_async_to_lds_b128 %0, %1, off"
                 :: "v"(lds_byte_addr),
                    "v"((unsigned long long)(uintptr_t)gaddr)
                 : "memory");
}
__device__ __forceinline__ void wait_async0() {
    asm volatile("s_wait_asynccnt 0x0" ::: "memory");
}
// Low 32 bits of a generic shared-memory pointer == wave-relative LDS address.
__device__ __forceinline__ unsigned lds_addr_of(const void* p) {
    return (unsigned)(uintptr_t)p;
}

// ---------------------------------------------------------------------------
// Kernel 1: row softmax of s[T,J] -> bf16 A[T,J].   One 256-thread block/row.
// ---------------------------------------------------------------------------
__global__ __launch_bounds__(256) void softmax_bf16_kernel(
        const float* __restrict__ s, unsigned short* __restrict__ A) {
    __shared__ float red[8];
    const int t   = blockIdx.x;
    const int tid = threadIdx.x;
    const int wave = tid >> 5;
    const int lane = tid & 31;

    const float* row = s + (size_t)t * J_DIM;
    float4 a = ((const float4*)row)[tid * 2 + 0];
    float4 b = ((const float4*)row)[tid * 2 + 1];
    float v[8] = {a.x, a.y, a.z, a.w, b.x, b.y, b.z, b.w};

    float m = v[0];
#pragma unroll
    for (int i = 1; i < 8; ++i) m = fmaxf(m, v[i]);
#pragma unroll
    for (int o = 16; o > 0; o >>= 1) m = fmaxf(m, __shfl_xor(m, o, 32));
    if (lane == 0) red[wave] = m;
    __syncthreads();
    m = red[0];
#pragma unroll
    for (int i = 1; i < 8; ++i) m = fmaxf(m, red[i]);

    float sum = 0.f;
#pragma unroll
    for (int i = 0; i < 8; ++i) { v[i] = __expf(v[i] - m); sum += v[i]; }
#pragma unroll
    for (int o = 16; o > 0; o >>= 1) sum += __shfl_xor(sum, o, 32);
    __syncthreads();
    if (lane == 0) red[wave] = sum;
    __syncthreads();
    sum = 0.f;
#pragma unroll
    for (int i = 0; i < 8; ++i) sum += red[i];
    const float scale = 1.0f / sum;

    unsigned int p[4];
#pragma unroll
    for (int i = 0; i < 4; ++i) {
        unsigned int lo = f32_to_bf16_bits(v[2 * i + 0] * scale);
        unsigned int hi = f32_to_bf16_bits(v[2 * i + 1] * scale);
        p[i] = lo | (hi << 16);
    }
    uint4 o4; o4.x = p[0]; o4.y = p[1]; o4.z = p[2]; o4.w = p[3];
    ((uint4*)(A + (size_t)t * J_DIM))[tid] = o4;
}

// ---------------------------------------------------------------------------
// Kernel 2: u[J,D] f32 -> Ut[D,J] bf16 (tiled 32x32 LDS transpose).
// ---------------------------------------------------------------------------
__global__ __launch_bounds__(256) void cvt_transpose_u_kernel(
        const float* __restrict__ u, unsigned short* __restrict__ ut) {
    __shared__ unsigned short tile[32][40];
    const int tid = threadIdx.x;
    const int d0 = blockIdx.x * 32;
    const int j0 = blockIdx.y * 32;

    {   // coalesced read along d
        int lj = tid >> 3;
        int ld = (tid & 7) * 4;
        float4 f = *(const float4*)(u + (size_t)(j0 + lj) * D_DIM + d0 + ld);
        tile[ld + 0][lj] = (unsigned short)f32_to_bf16_bits(f.x);
        tile[ld + 1][lj] = (unsigned short)f32_to_bf16_bits(f.y);
        tile[ld + 2][lj] = (unsigned short)f32_to_bf16_bits(f.z);
        tile[ld + 3][lj] = (unsigned short)f32_to_bf16_bits(f.w);
    }
    __syncthreads();
    {   // coalesced write along j
        int ld = tid >> 3;
        int lj = (tid & 7) * 4;
        uint2 o;
        o.x = (unsigned int)tile[ld][lj + 0] | ((unsigned int)tile[ld][lj + 1] << 16);
        o.y = (unsigned int)tile[ld][lj + 2] | ((unsigned int)tile[ld][lj + 3] << 16);
        *(uint2*)(ut + (size_t)(d0 + ld) * J_DIM + j0 + lj) = o;
    }
}

// ---------------------------------------------------------------------------
// Kernel 3: WMMA GEMM  C[t,d] = sum_j A[t,j]*Ut[d,j], stored as out[d*T + t].
// Async global->LDS staging (ASYNCcnt), ping-pong LDS, 1 barrier / K step.
// ---------------------------------------------------------------------------
__global__ __launch_bounds__(256) void c2q_wmma_kernel(
        const unsigned short* __restrict__ Abf,   // [T, J] bf16
        const unsigned short* __restrict__ Ut,    // [D, J] bf16 (transposed u)
        float* __restrict__ out) {                // [D, T] f32
    __shared__ __align__(16) unsigned short lA[2][BM * LDA];  // [row][k]
    __shared__ __align__(16) unsigned short lB[2][BN * LDB];  // [col][k]

    const int tid  = threadIdx.x;
    const int wave = tid >> 5;
    const int lane = tid & 31;
    const int wm   = wave & 3;     // 0..3 -> 32 rows each
    const int wn   = wave >> 2;    // 0..1 -> 64 cols each
    const int m0   = blockIdx.x * BM;
    const int n0   = blockIdx.y * BN;

    v8f acc[2][4];
#pragma unroll
    for (int i = 0; i < 2; ++i)
#pragma unroll
        for (int j = 0; j < 4; ++j)
            acc[i][j] = (v8f){0.f, 0.f, 0.f, 0.f, 0.f, 0.f, 0.f, 0.f};

    const int lrow = lane & 15;
    const int sel  = lane >> 4;    // half-wave select

    // Issue async copies of one K-tile (A: 128x32, B: 128x32; 2x16B per
    // thread per matrix, 512 chunks each).
    auto stage = [&](int kk, int buf) {
#pragma unroll
        for (int i = 0; i < 2; ++i) {
            int c   = tid * 2 + i;
            int row = c >> 2;
            int ko  = (c & 3) * 8;
            async_copy_b128(lds_addr_of(&lA[buf][row * LDA + ko]),
                            Abf + (size_t)(m0 + row) * J_DIM + kk + ko);
            async_copy_b128(lds_addr_of(&lB[buf][row * LDB + ko]),
                            Ut + (size_t)(n0 + row) * J_DIM + kk + ko);
        }
    };

    stage(0, 0);                       // prologue

    for (int kk = 0; kk < J_DIM; kk += BK) {
        const int buf = (kk >> 5) & 1;
        wait_async0();                 // my tile-DMA done
        __syncthreads();               // tile published workgroup-wide

        if (kk + BK < J_DIM) {
            stage(kk + BK, buf ^ 1);   // issue-only: overlaps WMMAs below
            if (kk + 2 * BK < J_DIM) {
                __builtin_prefetch(Abf + (size_t)(m0 + (tid >> 1)) * J_DIM + kk + 2 * BK, 0, 3);
                __builtin_prefetch(Ut  + (size_t)(n0 + (tid >> 1)) * J_DIM + kk + 2 * BK, 0, 3);
            }
        }

        // ---- fragments per ISA 16-bit layouts ----
        union Frag { uint4 q[2]; v16bf v; };
        Frag af[2], bf[4];
#pragma unroll
        for (int i = 0; i < 2; ++i) {
            int r = wm * 32 + i * 16 + lrow;
            // lane<16: K{0..7,16..23}; lane>=16: K{8..15,24..31}
            af[i].q[0] = *(const uint4*)(&lA[buf][r * LDA + sel * 8]);
            af[i].q[1] = *(const uint4*)(&lA[buf][r * LDA + 16 + sel * 8]);
        }
#pragma unroll
        for (int j = 0; j < 4; ++j) {
            int col = wn * 64 + j * 16 + lrow;
            int kb  = sel * 8;         // lane<16: K{0..7,16..23}; lane>=16: K{8..15,24..31}
            bf[j].q[0] = *(const uint4*)(&lB[buf][col * LDB + kb]);
            bf[j].q[1] = *(const uint4*)(&lB[buf][col * LDB + kb + 16]);
        }

#pragma unroll
        for (int i = 0; i < 2; ++i)
#pragma unroll
            for (int j = 0; j < 4; ++j)
                acc[i][j] = __builtin_amdgcn_wmma_f32_16x16x32_bf16(
                    false, af[i].v, false, bf[j].v,
                    (short)0, acc[i][j], false, false);
    }

    // ---- transposed store: lane owns column d, 8 consecutive t values ----
#pragma unroll
    for (int i = 0; i < 2; ++i) {
#pragma unroll
        for (int j = 0; j < 4; ++j) {
            int mb  = m0 + wm * 32 + i * 16;
            int nb  = n0 + wn * 64 + j * 16;
            int col = nb + lrow;               // d index
            int r0  = mb + sel * 8;            // t index (M = r or r+8)
            float* p = out + (size_t)col * T_DIM + r0;
            float4 lo, hi;
            lo.x = acc[i][j][0]; lo.y = acc[i][j][1];
            lo.z = acc[i][j][2]; lo.w = acc[i][j][3];
            hi.x = acc[i][j][4]; hi.y = acc[i][j][5];
            hi.z = acc[i][j][6]; hi.w = acc[i][j][7];
            *(float4*)(p + 0) = lo;
            *(float4*)(p + 4) = hi;
        }
    }
}

// ---------------------------------------------------------------------------
extern "C" void kernel_launch(void* const* d_in, const int* in_sizes, int n_in,
                              void* d_out, int out_size, void* d_ws, size_t ws_size,
                              hipStream_t stream) {
    const float* u = (const float*)d_in[0];   // [1, J, D]
    const float* s = (const float*)d_in[1];   // [T, J]
    float* out = (float*)d_out;               // [D, T]

    unsigned short* Abf = (unsigned short*)d_ws;               // 32 MB
    unsigned short* Ut  = Abf + (size_t)T_DIM * J_DIM;         // + 8 MB

    softmax_bf16_kernel<<<T_DIM, 256, 0, stream>>>(s, Abf);
    dim3 tgrid(D_DIM / 32, J_DIM / 32);
    cvt_transpose_u_kernel<<<tgrid, 256, 0, stream>>>(u, Ut);
    dim3 grid(T_DIM / BM, D_DIM / BN);
    c2q_wmma_kernel<<<grid, 256, 0, stream>>>(Abf, Ut, out);
}
